// PriceLSTMRegressor_53231824666782
// MI455X (gfx1250) — matrix operand order repaired and checked by reading, not compiled, not run on hardware
//
#include <hip/hip_runtime.h>
#include <hip/hip_bf16.h>

// ---------------------------------------------------------------------------
// PriceLSTMRegressor on gfx1250 (CDNA5 / MI455X)
//   B=1024, T=512, D=64, H=256, 2 LSTM layers + tanh FC head.
//
//  * Batch rows independent through recurrence -> 64 persistent workgroups,
//    each owning 16 batch rows for all 512 timesteps.
//  * Per step: gates[16,1024] = [x_t | h] (bf16 LDS) x Wcat^T (bf16, L2)
//    via v_wmma_f32_16x16x32_bf16. Each of 8 waves owns 32 hidden columns
//    and its i/f/g/o tiles -> LSTM cell update fully in registers.
//  * K-loop kept at unroll 1 + sched_barrier: per-iteration working set
//    (A frag + 8 B frags + 8 accs + cell state) stays in VGPRs, no spills.
//  * Nonlinearities use single-instruction transcendentals (v_exp_f32 /
//    v_rcp_f32 / v_tanh_f32) instead of branchy libm expf/tanhf -- these
//    are TRANS ops that co-execute with the WMMA pipe on CDNA5.
// ---------------------------------------------------------------------------

typedef __attribute__((ext_vector_type(16))) __bf16 v16bf;
typedef __attribute__((ext_vector_type(8)))  float  v8f;
typedef __attribute__((ext_vector_type(4)))  unsigned int v4u;

#define BM      16      // batch rows per workgroup
#define HDIM    256
#define TSTEPS  512
#define DDIM    64
#define A0S     328     // a0 row stride (elems): 64 x-cols + 256 h-cols + pad
#define A1S     520     // a1 row stride (elems): 256 h0 + 256 h1 + pad
#define K0      320     // layer-0 GEMM K = D + H
#define K1      512     // layer-1 GEMM K = H + H

#define LOG2E   1.4426950408889634f

// Branch-free single-instruction transcendentals (v_exp_f32 = 2^x, v_rcp_f32).
__device__ __forceinline__ float fast_sigmoid(float x) {
    return __builtin_amdgcn_rcpf(1.0f + __builtin_amdgcn_exp2f(-LOG2E * x));
}

__device__ __forceinline__ float fast_tanh(float x) {
#if __has_builtin(__builtin_amdgcn_tanhf)
    return __builtin_amdgcn_tanhf(x);            // v_tanh_f32
#elif __has_builtin(__builtin_amdgcn_tanh_f32)
    return __builtin_amdgcn_tanh_f32(x);
#else
    // tanh(x) = (1 - e^{-2x}) / (1 + e^{-2x}), e^{-2x} = 2^{-2*log2e*x}
    float t = __builtin_amdgcn_exp2f(-2.0f * LOG2E * x);
    return (1.0f - t) * __builtin_amdgcn_rcpf(1.0f + t);
#endif
}

// A-fragment (16x32 bf16) from LDS, per ISA 16-bit A layout:
// lanes 0-15: row = lane, K = {k0..k0+7} then {k0+16..k0+23}
// lanes 16-31: row = lane-16, K = {k0+8..k0+15} then {k0+24..k0+31}
__device__ __forceinline__ v16bf load_a_frag(const __bf16* base, int stride,
                                             int lane, int k0) {
    int row = lane & 15;
    int kh  = (lane >> 4) * 8;
    const __bf16* p = base + row * stride + k0 + kh;
    union { v4u u[2]; v16bf v; } un;
    un.u[0] = *(const v4u*)(p);        // 8 bf16 = 16B -> ds_load_b128
    un.u[1] = *(const v4u*)(p + 16);
    return un.v;
}

// B-fragment (32x16 bf16) from global. W row-major [N][K] (B = W^T).
// lanes 0-15: K={k0..k0+7} then {k0+16..k0+23} of column (n0+lane),
// lanes 16-31: K={k0+8..k0+15} then {k0+24..k0+31} of column (n0+lane-16).
__device__ __forceinline__ v16bf load_b_frag(const __bf16* __restrict__ W,
                                             int ldk, int n0, int lane, int k0) {
    int col = lane & 15;
    int kh  = (lane >> 4) * 8;
    const __bf16* p = W + (size_t)(n0 + col) * ldk + k0 + kh;
    union { v4u u[2]; v16bf v; } un;
    un.u[0] = *(const v4u*)(p);        // global_load_b128
    un.u[1] = *(const v4u*)(p + 16);
    return un.v;
}

// gates[16, 4H] GEMM for this wave's 8 tiles (4 gates x 2 subtiles at
// columns jb..jb+31 within each gate's H-range).
__device__ __forceinline__ void gemm_gates(const __bf16* __restrict__ W, int K,
                                           const __bf16* A, int astride,
                                           int lane, int jb, v8f acc[4][2]) {
#pragma unroll 1
    for (int k0 = 0; k0 < K; k0 += 32) {
        v16bf a = load_a_frag(A, astride, lane, k0);
#pragma unroll
        for (int g = 0; g < 4; ++g) {
#pragma unroll
            for (int s = 0; s < 2; ++s) {
                v16bf b = load_b_frag(W, K, g * HDIM + jb + s * 16, lane, k0);
                acc[g][s] = __builtin_amdgcn_wmma_f32_16x16x32_bf16(
                    false, a, false, b, (short)0, acc[g][s], false, false);
            }
        }
#if __has_builtin(__builtin_amdgcn_sched_barrier)
        __builtin_amdgcn_sched_barrier(0);   // keep per-iteration working set
#endif
    }
}

// Elementwise LSTM cell on this wave's 32 hidden columns; c state in regs,
// writes new h (bf16) into up to two LDS destinations. Branch-free TRANS ops.
__device__ __forceinline__ void lstm_cell(v8f acc[4][2], const float bias[4][2],
                                          v8f cst[2], int lane, int jb,
                                          __bf16* d0, int s0, int o0,
                                          __bf16* d1, int s1, int o1) {
    int colb = jb + (lane & 15);
    int rowb = (lane >> 4) * 8;
#pragma unroll
    for (int s = 0; s < 2; ++s) {
#pragma unroll
        for (int r = 0; r < 8; ++r) {
            float iv = fast_sigmoid(acc[0][s][r] + bias[0][s]);
            float fv = fast_sigmoid(acc[1][s][r] + bias[1][s]);
            float gv = fast_tanh   (acc[2][s][r] + bias[2][s]);
            float ov = fast_sigmoid(acc[3][s][r] + bias[3][s]);
            float c  = fv * cst[s][r] + iv * gv;
            cst[s][r] = c;
            float h  = ov * fast_tanh(c);
            int row = rowb + r;
            int col = colb + s * 16;
            d0[row * s0 + o0 + col] = (__bf16)h;
            if (d1) d1[row * s1 + o1 + col] = (__bf16)h;
        }
    }
}

// ---------------------------------------------------------------------------
// Prologue: repack fp32 weights into bf16 concatenated matrices + fused bias.
// ---------------------------------------------------------------------------
__global__ void pack_weights(const float* __restrict__ Wih0, const float* __restrict__ Whh0,
                             const float* __restrict__ bih0, const float* __restrict__ bhh0,
                             const float* __restrict__ Wih1, const float* __restrict__ Whh1,
                             const float* __restrict__ bih1, const float* __restrict__ bhh1,
                             __bf16* __restrict__ Wcat0, __bf16* __restrict__ Wcat1,
                             float* __restrict__ bsum0, float* __restrict__ bsum1) {
    int idx = blockIdx.x * blockDim.x + threadIdx.x;
    if (idx < 1024 * K0) {              // Wcat0: [1024][320] = [Wih0 | Whh0]
        int n = idx / K0, k = idx % K0;
        float v = (k < DDIM) ? Wih0[n * DDIM + k] : Whh0[n * HDIM + (k - DDIM)];
        Wcat0[idx] = (__bf16)v;
    }
    if (idx < 1024 * K1) {              // Wcat1: [1024][512] = [Wih1 | Whh1]
        int n = idx / K1, k = idx % K1;
        float v = (k < HDIM) ? Wih1[n * HDIM + k] : Whh1[n * HDIM + (k - HDIM)];
        Wcat1[idx] = (__bf16)v;
    }
    if (idx < 1024) {
        bsum0[idx] = bih0[idx] + bhh0[idx];
        bsum1[idx] = bih1[idx] + bhh1[idx];
    }
}

// ---------------------------------------------------------------------------
// Persistent fused 2-layer LSTM + FC head. One workgroup per 16 batch rows.
// 8 waves = 2 waves/SIMD on a WGP -> allow half the RF per wave.
// ---------------------------------------------------------------------------
__global__ __launch_bounds__(256)
__attribute__((amdgpu_waves_per_eu(2)))
void lstm_persistent(
    const float* __restrict__ x,
    const __bf16* __restrict__ Wcat0, const __bf16* __restrict__ Wcat1,
    const float* __restrict__ bsum0, const float* __restrict__ bsum1,
    const float* __restrict__ fc_w, const float* __restrict__ fc_b,
    float* __restrict__ out) {

    __shared__ __bf16 a0[BM * A0S];   // [x_t(64) | h0(256)] per row, padded
    __shared__ __bf16 a1[BM * A1S];   // [h0(256) | h1(256)] per row, padded

    const int tid  = threadIdx.x;
    const int lane = tid & 31;
    const int wv   = tid >> 5;        // 8 waves
    const int jb   = wv * 32;         // this wave's hidden-column base
    const int b0   = blockIdx.x * BM; // batch-row base

    // zero-init LDS state (h0 = h1 = 0)
    for (int i = tid; i < BM * A0S; i += 256) a0[i] = (__bf16)0.0f;
    for (int i = tid; i < BM * A1S; i += 256) a1[i] = (__bf16)0.0f;

    // hoist bias loads (column-dependent only)
    float bias0[4][2], bias1[4][2];
#pragma unroll
    for (int g = 0; g < 4; ++g)
#pragma unroll
        for (int s = 0; s < 2; ++s) {
            int col = g * HDIM + jb + s * 16 + (lane & 15);
            bias0[g][s] = bsum0[col];
            bias1[g][s] = bsum1[col];
        }

    v8f c0[2] = {};   // layer-0 cell state (registers)
    v8f c1[2] = {};   // layer-1 cell state (registers)
    __syncthreads();

#pragma unroll 1
    for (int t = 0; t < TSTEPS; ++t) {
        // stage x_t tile -> a0[:, 0:64] as bf16 (coalesced float4 loads)
        {
            int row = tid >> 4, seg = tid & 15;
            const float4 v = *(const float4*)(
                x + (((size_t)(b0 + row)) * TSTEPS + t) * DDIM + seg * 4);
            __bf16* d = a0 + row * A0S + seg * 4;
            d[0] = (__bf16)v.x; d[1] = (__bf16)v.y;
            d[2] = (__bf16)v.z; d[3] = (__bf16)v.w;
        }
        __syncthreads();

        // ---- layer 0: gates = [x_t | h0] @ Wcat0^T ----
        v8f acc0[4][2] = {};
        gemm_gates(Wcat0, K0, a0, A0S, lane, jb, acc0);
        __syncthreads();
        // h0 -> a0 h-region (for t+1) and a1 input region (for layer 1 now)
        lstm_cell(acc0, bias0, c0, lane, jb, a0, A0S, DDIM, a1, A1S, 0);
        __syncthreads();

        // ---- layer 1: gates = [h0 | h1] @ Wcat1^T ----
        v8f acc1[4][2] = {};
        gemm_gates(Wcat1, K1, a1, A1S, lane, jb, acc1);
        __syncthreads();
        lstm_cell(acc1, bias1, c1, lane, jb, a1, A1S, HDIM, (__bf16*)nullptr, 0, 0);
        __syncthreads();
    }

    // ---- FC head: out[b] = tanh(h1[b] . fc_w + fc_b), 16 rows per WG ----
    if (tid < BM) {
        float s = fc_b[0];
        const __bf16* h1 = a1 + tid * A1S + HDIM;
        for (int k = 0; k < HDIM; ++k) s += (float)h1[k] * fc_w[k];
        out[b0 + tid] = fast_tanh(s);
    }
}

// ---------------------------------------------------------------------------
extern "C" void kernel_launch(void* const* d_in, const int* in_sizes, int n_in,
                              void* d_out, int out_size, void* d_ws, size_t ws_size,
                              hipStream_t stream) {
    const float* x    = (const float*)d_in[0];
    const float* Wih0 = (const float*)d_in[1];
    const float* Whh0 = (const float*)d_in[2];
    const float* bih0 = (const float*)d_in[3];
    const float* bhh0 = (const float*)d_in[4];
    const float* Wih1 = (const float*)d_in[5];
    const float* Whh1 = (const float*)d_in[6];
    const float* bih1 = (const float*)d_in[7];
    const float* bhh1 = (const float*)d_in[8];
    const float* fcw  = (const float*)d_in[9];
    const float* fcb  = (const float*)d_in[10];

    // workspace layout (bytes)
    char*   ws    = (char*)d_ws;
    __bf16* Wcat0 = (__bf16*)(ws);                        // 1024*320*2 = 655360
    __bf16* Wcat1 = (__bf16*)(ws + 655360);               // 1024*512*2 = 1048576
    float*  bsum0 = (float*) (ws + 655360 + 1048576);     // 4096
    float*  bsum1 = (float*) (ws + 655360 + 1048576 + 4096);

    // repack weights to bf16 (covers max(1024*512) elems)
    pack_weights<<<(1024 * K1 + 255) / 256, 256, 0, stream>>>(
        Wih0, Whh0, bih0, bhh0, Wih1, Whh1, bih1, bhh1,
        Wcat0, Wcat1, bsum0, bsum1);

    // 1024 batch rows / 16 per WG = 64 persistent workgroups of 8 waves
    lstm_persistent<<<1024 / BM, 256, 0, stream>>>(
        x, Wcat0, Wcat1, bsum0, bsum1, fcw, fcb, (float*)d_out);
}
